// ARCLLMUnified_83923660964343
// MI455X (gfx1250) — compile-verified, matrix operand order checked by reading, not run on hardware
//
#include <hip/hip_runtime.h>
#include <hip/hip_bf16.h>

// ---------------------------------------------------------------------------
// Geodesic-attention transformer forward for MI455X (gfx1250, wave32, WMMA).
// GEMMs: v_wmma_f32_16x16x32_f16 fed from 64x64 LDS panels staged with
// double-buffered global_load_async_to_lds_b128 (ASYNCcnt) — the CDNA5 path.
// Weights are transcoded fp32 -> f16 (transposed to N x K) every call.
// ---------------------------------------------------------------------------

#define SEQ   512
#define DIM   512
#define NH    8
#define RR    16
#define HDIM  64
#define VOCAB 32000
#define NLAYER 2
#define FFD   2048
#define INV_SQRT_HD 0.125f   // 1/sqrt(64)

typedef _Float16 half_t;
typedef __attribute__((ext_vector_type(16))) _Float16 v16h;
typedef __attribute__((ext_vector_type(8)))  _Float16 v8h;
typedef __attribute__((ext_vector_type(8)))  float    v8f;

// ---- CDNA5 async global->LDS copy (ASYNCcnt) --------------------------------
__device__ __forceinline__ void async_b128_to_lds(const void* gptr, void* lds_ptr) {
  // LDS byte offset = low 32 bits of the flat shared-aperture address.
  unsigned int laddr = (unsigned int)(unsigned long long)(uintptr_t)lds_ptr;
  unsigned long long gaddr = (unsigned long long)(uintptr_t)gptr;
  asm volatile("global_load_async_to_lds_b128 %0, %1, off"
               :: "v"(laddr), "v"(gaddr) : "memory");
}
__device__ __forceinline__ void wait_async_zero() {
  asm volatile("s_wait_asynccnt 0x0" ::: "memory");
}

// ---- fast transcendentals (native gfx1250 TRANS ops when available) --------
__device__ __forceinline__ float fast_exp(float x) {
#if __has_builtin(__builtin_amdgcn_exp2f)
  return __builtin_amdgcn_exp2f(x * 1.44269504088896f);
#else
  return expf(x);
#endif
}
__device__ __forceinline__ float fast_tanh(float x) {
#if __has_builtin(__builtin_amdgcn_tanhf)
  return __builtin_amdgcn_tanhf(x);
#else
  return tanhf(x);
#endif
}

// ---------------------------------------------------------------- embedding
__global__ void k_embed(const int* __restrict__ ids,
                        const float* __restrict__ embed,
                        float* __restrict__ x) {
  int s = blockIdx.x;
  int row = ids[s];
  const float4* src = (const float4*)(embed + (size_t)row * DIM);
  float4* dst = (float4*)(x + (size_t)s * DIM);
  for (int i = threadIdx.x; i < DIM / 4; i += blockDim.x) dst[i] = src[i];
}

// ---------------------------------------------------------------- layernorm
__global__ __launch_bounds__(256)
void k_layernorm(const float* __restrict__ x,
                 const float* __restrict__ g, const float* __restrict__ b,
                 float* __restrict__ y, half_t* __restrict__ yh) {
  int s = blockIdx.x, tid = threadIdx.x;
  const float* xr = x + (size_t)s * DIM;
  __shared__ float r1[256], r2[256];
  float sum = 0.f, sq = 0.f;
  for (int d = tid; d < DIM; d += 256) { float v = xr[d]; sum += v; sq += v * v; }
  r1[tid] = sum; r2[tid] = sq; __syncthreads();
  for (int o = 128; o > 0; o >>= 1) {
    if (tid < o) { r1[tid] += r1[tid + o]; r2[tid] += r2[tid + o]; }
    __syncthreads();
  }
  float m   = r1[0] * (1.0f / DIM);
  float var = r2[0] * (1.0f / DIM) - m * m;
  float inv = rsqrtf(var + 1e-5f);
  for (int d = tid; d < DIM; d += 256) {
    float v = (xr[d] - m) * inv * g[d] + b[d];
    y[(size_t)s * DIM + d] = v;
    yh[(size_t)s * DIM + d] = (half_t)v;
  }
}

// ------------------------------------------- tiled fp32 -> f16 transpose
// dst (cols x rows) f16  =  transpose of src (rows x cols) f32
__global__ void k_transpose_f32_to_f16(const float* __restrict__ src,
                                       half_t* __restrict__ dst,
                                       int rows, int cols) {
  __shared__ float tile[32][33];
  int c0 = blockIdx.x * 32, r0 = blockIdx.y * 32;
  for (int i = threadIdx.y; i < 32; i += 8)
    tile[i][threadIdx.x] = src[(size_t)(r0 + i) * cols + c0 + threadIdx.x];
  __syncthreads();
  for (int i = threadIdx.y; i < 32; i += 8)
    dst[(size_t)(c0 + i) * rows + r0 + threadIdx.x] = (half_t)tile[threadIdx.x][i];
}

// ---------------------------------------------------------- WMMA fragments
// 16x32 f16 fragment from a 64-wide row-major LDS panel:
// lanes 0-15 hold row (row0+l), K = {k0..k0+7, k0+16..k0+23};
// lanes 16-31 hold row (row0+l), K = {k0+8..k0+15, k0+24..k0+31}.
__device__ __forceinline__ v16h frag_from_lds(const half_t* __restrict__ base,
                                              int row0, int k0, int lane) {
  int l  = lane & 15;
  int hi = (lane >> 4) & 1;
  const half_t* p = base + (row0 + l) * 64 + k0 + hi * 8;
  v8h a = *(const v8h*)(p);
  v8h b = *(const v8h*)(p + 16);
  v16h r;
#pragma unroll
  for (int i = 0; i < 8; i++) { r[i] = a[i]; r[i + 8] = b[i]; }
  return r;
}

#define WMMA_F16(acc, a, b) \
  __builtin_amdgcn_wmma_f32_16x16x32_f16(false, (a), false, (b), (short)0, (acc), false, false)

// ------------------------------------------------------ batched WMMA GEMM
// C[z] (MxN, f32, ldc) = A[z] (MxK f16, lda) * Bt[z]^T (Bt is NxK f16, ldb)
//                         + bias (len N, optional)
// Block = 256 threads = 8 waves (4 M x 2 N), block tile 64x64, K-panel 64,
// each wave computes 16x32 (two accumulators), double-buffered async
// global->LDS staging (4 B128 async loads / thread / panel).
// All launched tiles are full (M mult of 64, N mult of 64, K mult of 64).
// ACT: 0 = none, 1 = exact GELU.  WRITE_H: also emit f16 mirror of C.
template <int ACT, bool WRITE_H>
__global__ __launch_bounds__(256)
void k_gemm_wmma(const half_t* __restrict__ A, int lda, size_t sA_,
                 const half_t* __restrict__ Bt, int ldb, size_t sB_,
                 const float* __restrict__ bias,
                 float* __restrict__ C, int ldc, size_t sC_,
                 half_t* __restrict__ Ch, int ldch, size_t sCh_,
                 int M, int N, int K) {
  (void)M; (void)N;
  int z = blockIdx.z;
  A  += (size_t)z * sA_;
  Bt += (size_t)z * sB_;
  C  += (size_t)z * sC_;
  if (WRITE_H) Ch += (size_t)z * sCh_;

  __shared__ alignas(16) half_t sA[2][64 * 64];   //  8 KB per buffer
  __shared__ alignas(16) half_t sB[2][64 * 64];   //  8 KB per buffer

  int tid  = threadIdx.x;
  int lane = tid & 31;
  int w  = tid >> 5;
  int wm = w >> 1, wn = w & 1;          // 8 waves: 4 (M) x 2 (N)
  int m0b = blockIdx.y * 64;
  int n0b = blockIdx.x * 64;

  // Per-thread staging addresses: 64x64 panel = 512 B128 chunks, 2 per thread
  // for A and 2 for B (rows srow and srow+32).
  int srow = tid >> 3;
  int sch  = (tid & 7) * 8;
  const half_t* gA0 = A  + (size_t)(m0b + srow)      * lda + sch;
  const half_t* gA1 = A  + (size_t)(m0b + srow + 32) * lda + sch;
  const half_t* gB0 = Bt + (size_t)(n0b + srow)      * ldb + sch;
  const half_t* gB1 = Bt + (size_t)(n0b + srow + 32) * ldb + sch;
  int lo = srow * 64 + sch;

  v8f acc0 = {}, acc1 = {};

  auto stage = [&](int kp, int bufi) {
    async_b128_to_lds(gA0 + kp, &sA[bufi][lo]);
    async_b128_to_lds(gA1 + kp, &sA[bufi][32 * 64 + lo]);
    async_b128_to_lds(gB0 + kp, &sB[bufi][lo]);
    async_b128_to_lds(gB1 + kp, &sB[bufi][32 * 64 + lo]);
  };
  auto compute = [&](int bufi) {
#pragma unroll
    for (int kk = 0; kk < 2; ++kk) {
      v16h a  = frag_from_lds(sA[bufi], wm * 16,      kk * 32, lane);
      v16h b0 = frag_from_lds(sB[bufi], wn * 32,      kk * 32, lane);
      v16h b1 = frag_from_lds(sB[bufi], wn * 32 + 16, kk * 32, lane);
      acc0 = WMMA_F16(acc0, a, b0);
      acc1 = WMMA_F16(acc1, a, b1);
    }
  };

  const int P = K >> 6;                 // K is a multiple of 64 everywhere
  stage(0, 0);
  wait_async_zero();
  __syncthreads();

  int buf = 0;
  for (int p = 0; p < P - 1; ++p) {     // steady state: no branches inside
    stage((p + 1) << 6, buf ^ 1);       // prefetch next panel (async)
    compute(buf);                       // ds_load_b128 -> v_wmma
    wait_async_zero();
    __syncthreads();
    buf ^= 1;
  }
  compute(buf);                         // last panel

  int l  = lane & 15;
  int hi = (lane >> 4) & 1;
  int m0   = m0b + wm * 16;
  int col0 = n0b + wn * 32 + l;
  int col1 = col0 + 16;
  float bv0 = bias ? bias[col0] : 0.0f;
  float bv1 = bias ? bias[col1] : 0.0f;
#pragma unroll
  for (int i = 0; i < 8; i++) {
    int row = m0 + hi * 8 + i;           // C layout: VGPR i -> M = i / i+8
    float v0 = acc0[i] + bv0;
    float v1 = acc1[i] + bv1;
    if (ACT == 1) {
      v0 = 0.5f * v0 * (1.0f + erff(v0 * 0.70710678118f));
      v1 = 0.5f * v1 * (1.0f + erff(v1 * 0.70710678118f));
    }
    C[(size_t)row * ldc + col0] = v0;
    C[(size_t)row * ldc + col1] = v1;
    if (WRITE_H) {
      Ch[(size_t)row * ldch + col0] = (half_t)v0;
      Ch[(size_t)row * ldch + col1] = (half_t)v1;
    }
  }
}

// ------------------------------------------- per-head metric projections
// Aout[h,s,r] = sum_d src[s, h*64+d] * Amat[h,r,d]
// Mout[h,s,r] = 0.5 * sum_d src[s, h*64+d] * Wm[h,r,d]
// sqn[h,s]    = ||src[s, h*64 : h*64+64]||^2
__global__ void k_headpre(const float* __restrict__ src,
                          const float* __restrict__ Amat,
                          const float* __restrict__ Wm,
                          float* __restrict__ Aout,
                          float* __restrict__ Mout,
                          float* __restrict__ sqn) {
  int s = blockIdx.x, h = blockIdx.y, lane = threadIdx.x;   // 32 threads
  const float* xr = src + (size_t)s * DIM + h * HDIM;
  if (lane < RR) {
    const float* ar = Amat + ((size_t)h * RR + lane) * HDIM;
    const float* wr = Wm   + ((size_t)h * RR + lane) * HDIM;
    float da = 0.f, dm = 0.f;
#pragma unroll 8
    for (int d = 0; d < HDIM; d++) { float v = xr[d]; da += v * ar[d]; dm += v * wr[d]; }
    Aout[((size_t)h * SEQ + s) * RR + lane] = da;
    Mout[((size_t)h * SEQ + s) * RR + lane] = 0.5f * dm;
  }
  float p = 0.f;
  for (int d = lane; d < HDIM; d += 32) { float v = xr[d]; p += v * v; }
#pragma unroll
  for (int o = 16; o > 0; o >>= 1) p += __shfl_xor(p, o, 32);
  if (lane == 0) sqn[(size_t)h * SEQ + s] = p;
}

// ---------------------- tanh-modulated geodesic distance + row softmax
// One block per (query i, head h); 512 threads (16 wave32) cover all keys j.
// Reductions: wave32 shfl trees + one 16-entry cross-wave combine.
__global__ __launch_bounds__(512)
void k_attn(const float* __restrict__ qk,   // (H,S,S) q.k^T
            const float* __restrict__ Aq, const float* __restrict__ Ak,
            const float* __restrict__ Mq, const float* __restrict__ Mk,
            const float* __restrict__ qq, const float* __restrict__ kk,
            const float* __restrict__ bm,       // (H,R)
            const float* __restrict__ log_lam,  // (H)
            half_t* __restrict__ attnh) {       // (H,S,S) f16 probs
  int i = blockIdx.x, h = blockIdx.y, j = threadIdx.x;
  __shared__ float wred[16];
  __shared__ float mqi[RR], aqi[RR], bms[RR];
  if (j < RR) {
    mqi[j] = Mq[((size_t)h * SEQ + i) * RR + j];
    aqi[j] = Aq[((size_t)h * SEQ + i) * RR + j];
    bms[j] = bm[h * RR + j];
  }
  __syncthreads();

  const float* mkj = Mk + ((size_t)h * SEQ + j) * RR;
  const float* akj = Ak + ((size_t)h * SEQ + j) * RR;
  float quad = 0.f;
#pragma unroll
  for (int r = 0; r < RR; r++) {
    float md = fast_tanh(mqi[r] + mkj[r] + bms[r]);   // v_tanh_f32 (TRANS)
    float pd = aqi[r] - akj[r];
    float t = md * pd;
    quad += t * t;
  }
  float lam = fast_exp(log_lam[h]);
  float sq  = qq[(size_t)h * SEQ + i] + kk[(size_t)h * SEQ + j]
            - 2.0f * qk[((size_t)h * SEQ + i) * SEQ + j];
  float dist = fminf(fmaxf(quad + lam * sq, 0.0f) * INV_SQRT_HD, 50.0f);
  float sc = -dist;

  // row max: wave shfl tree, then cross-wave combine
  float m = sc;
#pragma unroll
  for (int o = 16; o > 0; o >>= 1) m = fmaxf(m, __shfl_xor(m, o, 32));
  if ((j & 31) == 0) wred[j >> 5] = m;
  __syncthreads();
  float mx = wred[0];
#pragma unroll
  for (int t = 1; t < 16; t++) mx = fmaxf(mx, wred[t]);

  float e = fast_exp(sc - mx);          // v_exp_f32 (TRANS)
  float s = e;
#pragma unroll
  for (int o = 16; o > 0; o >>= 1) s += __shfl_xor(s, o, 32);
  __syncthreads();                      // wred reuse
  if ((j & 31) == 0) wred[j >> 5] = s;
  __syncthreads();
  float tot = 0.f;
#pragma unroll
  for (int t = 0; t < 16; t++) tot += wred[t];

  attnh[((size_t)h * SEQ + i) * SEQ + j] = (half_t)(e / tot);
}

// -------------------------------------------- parallel transport residual
__global__ __launch_bounds__(256)
void k_transport(float* __restrict__ x, const float* __restrict__ delta,
                 const float* __restrict__ scale_p, int layer) {
  int s = blockIdx.x, tid = threadIdx.x;
  __shared__ float r1[256], r2[256];
  float* xr = x + (size_t)s * DIM;
  const float* dr = delta + (size_t)s * DIM;
  float dot = 0.f, nrm = 0.f;
  for (int d = tid; d < DIM; d += 256) { float xv = xr[d]; dot += xv * dr[d]; nrm += xv * xv; }
  r1[tid] = dot; r2[tid] = nrm; __syncthreads();
  for (int o = 128; o > 0; o >>= 1) {
    if (tid < o) { r1[tid] += r1[tid + o]; r2[tid] += r2[tid + o]; }
    __syncthreads();
  }
  float ratio = r1[0] / (r2[0] + 1e-8f);
  float sc = scale_p[layer];
  for (int d = tid; d < DIM; d += 256) {
    float xv = xr[d];
    xr[d] = xv + sc * (dr[d] - ratio * xv);
  }
}

// ===========================================================================
extern "C" void kernel_launch(void* const* d_in, const int* in_sizes, int n_in,
                              void* d_out, int out_size, void* d_ws, size_t ws_size,
                              hipStream_t stream) {
  (void)in_sizes; (void)n_in; (void)out_size; (void)ws_size;

  const int*   ids     = (const int*)  d_in[0];
  const float* embed   = (const float*)d_in[1];
  const float* Wq      = (const float*)d_in[2];
  const float* bq      = (const float*)d_in[3];
  const float* Wk      = (const float*)d_in[4];
  const float* bk      = (const float*)d_in[5];
  const float* Wv      = (const float*)d_in[6];
  const float* bv      = (const float*)d_in[7];
  const float* Wo      = (const float*)d_in[8];
  const float* bo      = (const float*)d_in[9];
  const float* Amat    = (const float*)d_in[10];
  const float* log_lam = (const float*)d_in[11];
  const float* Wm      = (const float*)d_in[12];
  const float* bm      = (const float*)d_in[13];
  const float* ln1_g   = (const float*)d_in[14];
  const float* ln1_b   = (const float*)d_in[15];
  const float* ln2_g   = (const float*)d_in[16];
  const float* ln2_b   = (const float*)d_in[17];
  const float* scale   = (const float*)d_in[18];
  const float* Wfc1    = (const float*)d_in[19];
  const float* bfc1    = (const float*)d_in[20];
  const float* Wfc2    = (const float*)d_in[21];
  const float* bfc2    = (const float*)d_in[22];
  const float* lnf_g   = (const float*)d_in[23];
  const float* lnf_b   = (const float*)d_in[24];
  const float* Whead   = (const float*)d_in[25];
  const float* bhead   = (const float*)d_in[26];
  float* out = (float*)d_out;

  // ------------------------------------------------ workspace carving
  char* ws = (char*)d_ws;
  size_t off = 0;
  auto alloc = [&](size_t bytes) -> void* {
    void* p = ws + off;
    off = (off + bytes + 255) & ~(size_t)255;
    return p;
  };
  float*  x     = (float*) alloc((size_t)SEQ * DIM * 4);
  float*  xln   = (float*) alloc((size_t)SEQ * DIM * 4);
  half_t* xlnh  = (half_t*)alloc((size_t)SEQ * DIM * 2);
  float*  q     = (float*) alloc((size_t)SEQ * DIM * 4);
  float*  k     = (float*) alloc((size_t)SEQ * DIM * 4);
  float*  v     = (float*) alloc((size_t)SEQ * DIM * 4);
  half_t* qh    = (half_t*)alloc((size_t)SEQ * DIM * 2);
  half_t* kh    = (half_t*)alloc((size_t)SEQ * DIM * 2);
  half_t* vt    = (half_t*)alloc((size_t)DIM * SEQ * 2);   // (D, S) = v^T
  float*  qkb   = (float*) alloc((size_t)NH * SEQ * SEQ * 4);
  float*  Aqb   = (float*) alloc((size_t)NH * SEQ * RR * 4);
  float*  Akb   = (float*) alloc((size_t)NH * SEQ * RR * 4);
  float*  Mqb   = (float*) alloc((size_t)NH * SEQ * RR * 4);
  float*  Mkb   = (float*) alloc((size_t)NH * SEQ * RR * 4);
  float*  qqb   = (float*) alloc((size_t)NH * SEQ * 4);
  float*  kkb   = (float*) alloc((size_t)NH * SEQ * 4);
  half_t* attnp = (half_t*)alloc((size_t)NH * SEQ * SEQ * 2);
  float*  ao    = (float*) alloc((size_t)SEQ * DIM * 4);
  half_t* aoh   = (half_t*)alloc((size_t)SEQ * DIM * 2);
  float*  tmp   = (float*) alloc((size_t)SEQ * DIM * 4);
  float*  h1    = (float*) alloc((size_t)SEQ * FFD * 4);
  half_t* h1h   = (half_t*)alloc((size_t)SEQ * FFD * 2);
  half_t* wT    = (half_t*)alloc((size_t)VOCAB * DIM * 2); // reused for all W

  const dim3 tb(32, 8);
  auto tpose = [&](const float* srcp, int rows, int cols) {
    k_transpose_f32_to_f16<<<dim3(cols / 32, rows / 32), tb, 0, stream>>>(srcp, wT, rows, cols);
  };
  auto ggrid = [](int M, int N, int batch) { return dim3(N / 64, M / 64, batch); };

  // ------------------------------------------------ embedding
  k_embed<<<SEQ, 128, 0, stream>>>(ids, embed, x);

  for (int l = 0; l < NLAYER; l++) {
    const size_t wdd = (size_t)l * DIM * DIM;
    // ---- LN1
    k_layernorm<<<SEQ, 256, 0, stream>>>(x, ln1_g + l * DIM, ln1_b + l * DIM, xln, xlnh);
    // ---- Q, K, V projections (WMMA), f16 mirrors for q/k
    tpose(Wq + wdd, DIM, DIM);
    k_gemm_wmma<0, true><<<ggrid(SEQ, DIM, 1), 256, 0, stream>>>(
        xlnh, DIM, 0, wT, DIM, 0, bq + l * DIM, q, DIM, 0, qh, DIM, 0, SEQ, DIM, DIM);
    tpose(Wk + wdd, DIM, DIM);
    k_gemm_wmma<0, true><<<ggrid(SEQ, DIM, 1), 256, 0, stream>>>(
        xlnh, DIM, 0, wT, DIM, 0, bk + l * DIM, k, DIM, 0, kh, DIM, 0, SEQ, DIM, DIM);
    tpose(Wv + wdd, DIM, DIM);
    k_gemm_wmma<0, false><<<ggrid(SEQ, DIM, 1), 256, 0, stream>>>(
        xlnh, DIM, 0, wT, DIM, 0, bv + l * DIM, v, DIM, 0, (half_t*)nullptr, 0, 0, SEQ, DIM, DIM);
    // ---- v^T in f16 for attn@v
    k_transpose_f32_to_f16<<<dim3(DIM / 32, SEQ / 32), tb, 0, stream>>>(v, vt, SEQ, DIM);
    // ---- per-head metric projections
    const float* Al  = Amat + (size_t)l * NH * RR * HDIM;
    const float* Wml = Wm   + (size_t)l * NH * RR * HDIM;
    k_headpre<<<dim3(SEQ, NH), 32, 0, stream>>>(q, Al, Wml, Aqb, Mqb, qqb);
    k_headpre<<<dim3(SEQ, NH), 32, 0, stream>>>(k, Al, Wml, Akb, Mkb, kkb);
    // ---- batched q.k^T per head (A/Bt are column slices of qh/kh)
    k_gemm_wmma<0, false><<<ggrid(SEQ, SEQ, NH), 256, 0, stream>>>(
        qh, DIM, HDIM, kh, DIM, HDIM, (const float*)nullptr,
        qkb, SEQ, (size_t)SEQ * SEQ, (half_t*)nullptr, 0, 0, SEQ, SEQ, HDIM);
    // ---- tanh-metric distance + softmax -> f16 probs
    k_attn<<<dim3(SEQ, NH), 512, 0, stream>>>(
        qkb, Aqb, Akb, Mqb, Mkb, qqb, kkb,
        bm + (size_t)l * NH * RR, log_lam + (size_t)l * NH, attnp);
    // ---- batched attn @ v  -> ao columns [h*64, h*64+64)
    k_gemm_wmma<0, true><<<ggrid(SEQ, HDIM, NH), 256, 0, stream>>>(
        attnp, SEQ, (size_t)SEQ * SEQ, vt, SEQ, (size_t)HDIM * SEQ,
        (const float*)nullptr, ao, DIM, HDIM, aoh, DIM, HDIM, SEQ, HDIM, SEQ);
    // ---- output projection
    tpose(Wo + wdd, DIM, DIM);
    k_gemm_wmma<0, false><<<ggrid(SEQ, DIM, 1), 256, 0, stream>>>(
        aoh, DIM, 0, wT, DIM, 0, bo + l * DIM, tmp, DIM, 0, (half_t*)nullptr, 0, 0, SEQ, DIM, DIM);
    // ---- transport residual (attention)
    k_transport<<<SEQ, 256, 0, stream>>>(x, tmp, scale, l);
    // ---- LN2 + FFN
    k_layernorm<<<SEQ, 256, 0, stream>>>(x, ln2_g + l * DIM, ln2_b + l * DIM, xln, xlnh);
    tpose(Wfc1 + (size_t)l * DIM * FFD, DIM, FFD);
    k_gemm_wmma<1, true><<<ggrid(SEQ, FFD, 1), 256, 0, stream>>>(
        xlnh, DIM, 0, wT, DIM, 0, bfc1 + (size_t)l * FFD,
        h1, FFD, 0, h1h, FFD, 0, SEQ, FFD, DIM);
    tpose(Wfc2 + (size_t)l * FFD * DIM, FFD, DIM);
    k_gemm_wmma<0, false><<<ggrid(SEQ, DIM, 1), 256, 0, stream>>>(
        h1h, FFD, 0, wT, FFD, 0, bfc2 + l * DIM,
        tmp, DIM, 0, (half_t*)nullptr, 0, 0, SEQ, DIM, FFD);
    // ---- transport residual (FFN)
    k_transport<<<SEQ, 256, 0, stream>>>(x, tmp, scale, l);
  }

  // ------------------------------------------------ final LN + LM head
  k_layernorm<<<SEQ, 256, 0, stream>>>(x, lnf_g, lnf_b, xln, xlnh);
  tpose(Whead, DIM, VOCAB);                       // wT = Whead^T (V x D) f16
  k_gemm_wmma<0, false><<<ggrid(SEQ, VOCAB, 1), 256, 0, stream>>>(
      xlnh, DIM, 0, wT, DIM, 0, bhead, out, VOCAB, 0,
      (half_t*)nullptr, 0, 0, SEQ, VOCAB, DIM);
}
